// FeatureEmbed_20942260535631
// MI455X (gfx1250) — matrix operand order repaired and checked by reading, not compiled, not run on hardware
//
#include <hip/hip_runtime.h>

// ---------------- problem constants ----------------
#define BT 512
#define SQ 512
#define T  20
#define E  32
#define D  322          // 10*E + 2
#define M_TOTAL (BT * SQ)

#define BM 64           // rows per block (4 row-tiles of 16)
#define RT 4            // row tiles per block
#define KT 11           // K tiles of 32  (352 >= 322, zero padded)
#define NT 21           // N tiles of 16  (336 >= 322, zero padded)
#define NQ ((KT * 32) / 4)   // 88 quads per row

typedef float        v8f  __attribute__((ext_vector_type(8)));
typedef unsigned int v8u  __attribute__((ext_vector_type(8)));
typedef unsigned int v2u  __attribute__((ext_vector_type(2)));
typedef __bf16       v16bf __attribute__((ext_vector_type(16)));

// float -> bf16 round-to-nearest-even
static __device__ __forceinline__ unsigned short f2bf(float f) {
    unsigned u = __float_as_uint(f);
    u = (u + 0x7FFFu + ((u >> 16) & 1u)) >> 16;
    return (unsigned short)u;
}

static __device__ __forceinline__ float4 ld4(const float* p) {
    return *(const float4*)p;
}

// ---------------------------------------------------------------------------
// Prepack W (D x D row major; out[o] = sum_d final[d]*W[o,d]) into bf16
// B-matrix fragments: B[k][n] = W[n*D + k], tile (kt,nt), per-lane layout:
//   lane = kh*16 + nloc ; VGPR v holds K = kt*32 + 16*kh + 2v (+1 in hi half)
// One wave32 per fragment. 231 fragments * 1 KB = 236 KB in d_ws (L2 resident).
// ---------------------------------------------------------------------------
__global__ void prepack_W_kernel(const float* __restrict__ W,
                                 unsigned int* __restrict__ Wp)
{
    int frag = blockIdx.x;            // kt*NT + nt
    int kt = frag / NT, nt = frag % NT;
    int lane = threadIdx.x;           // 0..31
    int nloc = lane & 15, kh = lane >> 4;
    int n = nt * 16 + nloc;

    unsigned int tmp[8];
#pragma unroll
    for (int v = 0; v < 8; ++v) {
        int k0 = kt * 32 + 16 * kh + 2 * v;
        float f0 = (n < D && (k0    ) < D) ? W[n * D + k0    ] : 0.0f;
        float f1 = (n < D && (k0 + 1) < D) ? W[n * D + k0 + 1] : 0.0f;
        tmp[v] = (unsigned)f2bf(f0) | ((unsigned)f2bf(f1) << 16);
    }
    *(v8u*)&Wp[((long)frag * 32 + lane) * 8] = *(v8u*)tmp;
}

// ---------------------------------------------------------------------------
// Fused gather + GEMM + bias + leaky-ReLU.
// Gather is quad-vectorized: each 4-aligned feature quad maps to one even
// VGPR pair of the packed A fragment (same K-half), so one float4 load +
// one ds_store_b64 covers 4 elements. Task order is quad-major so the
// segment branch is wave-uniform.
// Each wave owns 3 N-tiles (nt, nt+8, nt+16) processed jointly so each
// A fragment (LDS) feeds 3 WMMAs, with B double-buffered from L2.
// ---------------------------------------------------------------------------
__global__ __launch_bounds__(256)
void feature_embed_gemm_kernel(const float* __restrict__ feature,
                               const float* __restrict__ jt,
                               const float* __restrict__ te,
                               const float* __restrict__ ce,
                               const float* __restrict__ bvec,
                               const unsigned int* __restrict__ Wp,
                               float* __restrict__ out)
{
    __shared__ float ldsFeat[BM * 12];                            // 3 KB
    __shared__ __align__(32) unsigned int ldsA[RT * KT * 32 * 8]; // 44 KB

    const int tid = threadIdx.x;
    const long rbase = (long)blockIdx.x * BM;

    // stage raw features for this block's 64 rows (streamed once -> NT loads)
    for (int i = tid; i < BM * 12; i += 256)
        ldsFeat[i] = __builtin_nontemporal_load(&feature[rbase * 12 + i]);
    __syncthreads();

    // ---- quad-vectorized gather into packed A fragments ----
    // (contiguous K order: element s of lane half h holds K = 16*h + s)
    for (int t = tid; t < NQ * BM; t += 256) {
        int row_in_blk = t & (BM - 1);     // quad-major: q uniform per wave
        int q  = t >> 6;
        int d0 = q * 4;
        long row = rbase + row_in_blk;
        int bt = (int)(row / SQ);
        const float* f12 = &ldsFeat[row_in_blk * 12];

        float4 vals = make_float4(0.f, 0.f, 0.f, 0.f);
        if (d0 < 32) {                                    // typeEmb (row0->0)
            int id = (int)f12[0];
            if (id != 0) vals = ld4(&te[id * E + d0]);
        } else if (d0 < 64) {                             // tableEmb
            int id = (int)f12[1];
            vals = ld4(&jt[((long)bt * T + id) * E + (d0 - 32)]);
        } else if (d0 < 128) {                            // columnsEmb (2x)
            int q2 = (d0 - 64) >> 5, e = (d0 - 64) & 31;
            int id = (int)f12[2 + q2];
            if (id != 0) vals = ld4(&ce[id * E + e]);
        } else if (d0 < 224) {                            // joinEmb (3x)
            int q2 = (d0 - 128) >> 5, e = (d0 - 128) & 31;
            int id = (int)f12[4 + q2];
            vals = ld4(&jt[((long)bt * T + id) * E + e]);
        } else if (d0 < 320) {                            // joinColumnsEmb (3x)
            int q2 = (d0 - 224) >> 5, e = (d0 - 224) & 31;
            int id = (int)f12[7 + q2];
            if (id != 0) vals = ld4(&ce[id * E + e]);
        } else if (d0 == 320) {                           // cost_card + pad
            vals.x = f12[10];
            vals.y = f12[11];
        }                                                 // else: K padding

        // fragment slot for this quad
        int kt  = d0 >> 5, kin = d0 & 31;
        int kh  = kin >> 4;
        int v0  = (kin & 15) >> 1; // even
        int m   = row_in_blk & 15, rt = row_in_blk >> 4;
        int lane = kh * 16 + m;
        unsigned int* dst =
            &ldsA[(((unsigned)rt * KT + kt) * 32 + lane) * 8 + v0];
        v2u packed;
        packed.x = (unsigned)f2bf(vals.x) | ((unsigned)f2bf(vals.y) << 16);
        packed.y = (unsigned)f2bf(vals.z) | ((unsigned)f2bf(vals.w) << 16);
        *(v2u*)dst = packed;                  // ds_store_b64, 8B aligned
    }
    __syncthreads();

    const int wid  = tid >> 5;        // 8 waves (wave32)
    const int lane = tid & 31;
    const int cn   = lane & 15;       // C/D: N = lane%16
    const int ch   = lane >> 4;       // C/D: M = v + 8*(lane/16)

    // this wave's three N tiles (invalid slots alias slot 0, guarded at store)
    const bool has1 = (wid + 8)  < NT;
    const bool has2 = (wid + 16) < NT;
    const int  nt0  = wid;
    const int  nt1  = has1 ? wid + 8  : wid;
    const int  nt2  = has2 ? wid + 16 : wid;

    auto loadB = [&](int kt, int nt) -> v8u {
        return *(const v8u*)&Wp[(((long)kt * NT + nt) * 32 + lane) * 8];
    };

    v8f c0[RT], c1[RT], c2[RT];
#pragma unroll
    for (int rt = 0; rt < RT; ++rt) { c0[rt] = (v8f){}; c1[rt] = (v8f){}; c2[rt] = (v8f){}; }

    // B double-buffer: issue kt+1 loads before consuming kt
    v8u b0 = loadB(0, nt0), b1 = loadB(0, nt1), b2 = loadB(0, nt2);
    for (int kt = 0; kt < KT; ++kt) {
        const int ktn = (kt + 1 < KT) ? kt + 1 : kt;
        v8u p0 = loadB(ktn, nt0);
        v8u p1 = loadB(ktn, nt1);
        v8u p2 = loadB(ktn, nt2);
        v16bf bf0 = __builtin_bit_cast(v16bf, b0);
        v16bf bf1 = __builtin_bit_cast(v16bf, b1);
        v16bf bf2 = __builtin_bit_cast(v16bf, b2);
#pragma unroll
        for (int rt = 0; rt < RT; ++rt) {
            v8u araw = *(const v8u*)&ldsA[(((unsigned)rt * KT + kt) * 32 + lane) * 8];
            v16bf af = __builtin_bit_cast(v16bf, araw);
            c0[rt] = __builtin_amdgcn_wmma_f32_16x16x32_bf16(
                false, af, false, bf0, (short)0, c0[rt], false, false);
            c1[rt] = __builtin_amdgcn_wmma_f32_16x16x32_bf16(
                false, af, false, bf1, (short)0, c1[rt], false, false);
            c2[rt] = __builtin_amdgcn_wmma_f32_16x16x32_bf16(
                false, af, false, bf2, (short)0, c2[rt], false, false);
        }
        b0 = p0; b1 = p1; b2 = p2;
    }

    // fused bias + leaky-ReLU stores (non-temporal: don't thrash L2)
    auto storeTile = [&](const v8f* c, int nt, bool valid) {
        if (!valid) return;
        int n = nt * 16 + cn;
        if (n >= D) return;
        float bb = bvec[n];
#pragma unroll
        for (int rt = 0; rt < RT; ++rt) {
#pragma unroll
            for (int v = 0; v < 8; ++v) {
                long row = rbase + rt * 16 + 8 * ch + v;
                float val = c[rt][v] + bb;
                val = val >= 0.0f ? val : 0.01f * val;
                __builtin_nontemporal_store(val, &out[row * D + n]);
            }
        }
    };
    storeTile(c0, nt0, true);
    storeTile(c1, nt1, has1);
    storeTile(c2, nt2, has2);
}

extern "C" void kernel_launch(void* const* d_in, const int* in_sizes, int n_in,
                              void* d_out, int out_size, void* d_ws, size_t ws_size,
                              hipStream_t stream)
{
    const float* feature = (const float*)d_in[0];   // (BT,SQ,12)
    const float* jt      = (const float*)d_in[1];   // (BT,T,E)
    const float* te      = (const float*)d_in[2];   // (TYPES,E)
    const float* ce      = (const float*)d_in[3];   // (COLS,E)
    const float* W       = (const float*)d_in[4];   // (D,D)
    const float* b       = (const float*)d_in[5];   // (D,)
    float* out = (float*)d_out;
    unsigned int* Wp = (unsigned int*)d_ws;         // 236,544 bytes used

    prepack_W_kernel<<<KT * NT, 32, 0, stream>>>(W, Wp);
    feature_embed_gemm_kernel<<<M_TOTAL / BM, 256, 0, stream>>>(
        feature, jt, te, ce, b, Wp, out);
}
